// GatedSlotAttention_13752485282234
// MI455X (gfx1250) — matrix operand order repaired and checked by reading, not compiled, not run on hardware
//
#include <hip/hip_runtime.h>

typedef __bf16 bf16;
typedef __attribute__((ext_vector_type(16))) __bf16 v16bf;
typedef __attribute__((ext_vector_type(8)))  __bf16 v8bf;
typedef __attribute__((ext_vector_type(8)))  float  v8f;

#define BB   2
#define TSEQ 2048
#define DIM  1024
#define HH   4
#define DKE  256      // head_k_dim == head_v_dim == num_slots == 256
#define CCH  32       // chunk length (== WMMA K-depth for bf16)

// ---------------- WMMA helpers (CDNA5 wave32, 16x16x32 bf16, f32 accum) --------

__device__ __forceinline__ v8f wmma_bf(v16bf a, v16bf b, v8f c) {
  return __builtin_amdgcn_wmma_f32_16x16x32_bf16(false, a, false, b, (short)0, c,
                                                 false, false);
}

// A-operand 16x32: lane row = lane%16; K-runs: lanes 0-15 -> {0..7,16..23},
// lanes 16-31 -> {8..15,24..31}
__device__ __forceinline__ v16bf load_a_f32(const float* rowbase, int ld, int k0) {
  const int lane = threadIdx.x & 31;
  const float* p = rowbase + (size_t)(lane & 15) * ld + k0 + ((lane >> 4) << 3);
  v16bf a;
#pragma unroll
  for (int i = 0; i < 8; ++i) { a[i] = (bf16)p[i]; a[i + 8] = (bf16)p[i + 16]; }
  return a;
}
__device__ __forceinline__ v16bf load_a_bf(const bf16* rowbase, int ld, int k0) {
  const int lane = threadIdx.x & 31;
  const bf16* p = rowbase + (size_t)(lane & 15) * ld + k0 + ((lane >> 4) << 3);
  v16bf a;
#pragma unroll
  for (int i = 0; i < 8; ++i) { a[i] = p[i]; a[i + 8] = p[i + 16]; }
  return a;
}
// B-operand 32x16: lane col = lane%16; K-runs: lanes 0-15 -> {0..7,16..23},
// lanes 16-31 -> {8..15,24..31}
__device__ __forceinline__ v16bf load_b_f32(const float* base, int ld, int n0, int k0) {
  const int lane = threadIdx.x & 31;
  const float* p = base + (size_t)(n0 + (lane & 15)) * ld + k0 + ((lane >> 4) << 3);
  v16bf b;
#pragma unroll
  for (int i = 0; i < 8; ++i) { b[i] = (bf16)p[i]; b[i + 8] = (bf16)p[i + 16]; }
  return b;
}
__device__ __forceinline__ v16bf load_b_bf(const bf16* base, int ld, int n0, int k0) {
  const int lane = threadIdx.x & 31;
  const bf16* p = base + (size_t)(n0 + (lane & 15)) * ld + k0 + ((lane >> 4) << 3);
  v16bf b;
#pragma unroll
  for (int i = 0; i < 8; ++i) { b[i] = p[i]; b[i + 8] = p[i + 16]; }
  return b;
}

__device__ __forceinline__ float silu_f(float x) { return x / (1.0f + expf(-x)); }

// ---------------- Kernel 1: fused QKVF projections + activations ---------------
__global__ __launch_bounds__(128) void gsa_proj(
    const float* __restrict__ hs,
    const float* __restrict__ Wq, const float* __restrict__ Wk,
    const float* __restrict__ Wv, const float* __restrict__ Wf,
    bf16* __restrict__ qb, bf16* __restrict__ kb, bf16* __restrict__ vb,
    float* __restrict__ flb, float* __restrict__ sb)
{
  const int tile = blockIdx.x * 4 + (threadIdx.x >> 5);
  const int rt = tile >> 8;            // 256 column tiles
  const int ct = tile & 255;
  const int rowBase = rt * 16;         // bt index
  const int colBase = ct * 16;         // concat(q,k,v,f) column
  const int proj = colBase >> 10;
  const int ncol0 = colBase & 1023;
  const float* W = (proj == 0) ? Wq : (proj == 1) ? Wk : (proj == 2) ? Wv : Wf;

  v8f acc = {};
#pragma unroll 4
  for (int kk = 0; kk < 32; ++kk) {
    v16bf a  = load_a_f32(hs + (size_t)rowBase * DIM, DIM, kk * 32);
    v16bf bm = load_b_f32(W, DIM, ncol0, kk * 32);
    acc = wmma_bf(a, bm, acc);
  }

  const int lane = threadIdx.x & 31;
  const int rowoff = (lane >> 4) << 3;
  const int jl = ncol0 + (lane & 15);
  const int h = jl >> 8, e2 = jl & 255;
#pragma unroll
  for (int r = 0; r < 8; ++r) {
    const int bt = rowBase + r + rowoff;
    const int b = bt >> 11, t = bt & 2047;
    const size_t idx = ((size_t)(b * HH + h) * TSEQ + (size_t)t) * DKE + e2;
    float x = acc[r];
    if (proj == 3) {
      float fl = (fminf(x, 0.0f) - log1pf(expf(-fabsf(x)))) * 0.125f;  // logsigmoid/8
      flb[idx] = fl;
      sb[idx] = 1.0f - expf(fl);
    } else {
      bf16 y = (bf16)silu_f(x);
      if (proj == 0) qb[idx] = y;
      else if (proj == 1) kb[idx] = y;
      else vb[idx] = y;
    }
  }
}

// ---------------- Kernel 2: pass-1 chunked GLA (state S[256 x 16] in accums) ---
__global__ __launch_bounds__(128) void gsa_pass1(
    const bf16* __restrict__ qg, const bf16* __restrict__ kg,
    const float* __restrict__ flg, const float* __restrict__ sg,
    float* __restrict__ okg)
{
  const int bh = blockIdx.x >> 4;
  const int mbase = (blockIdx.x & 15) * 16;
  const bf16* qh  = qg  + (size_t)bh * TSEQ * DKE;
  const bf16* kh  = kg  + (size_t)bh * TSEQ * DKE;
  const float* flh = flg + (size_t)bh * TSEQ * DKE + mbase;
  const float* sh  = sg  + (size_t)bh * TSEQ * DKE + mbase;
  float* okh = okg + (size_t)bh * TSEQ * DKE + mbase;

  __shared__ __align__(16) bf16 KcT[DKE * CCH];    // [dk][t]  16KB (transposed K)
  __shared__ __align__(16) bf16 S0c[16 * DKE];     // [m][dk]  scaled state, 8KB
  __shared__ __align__(16) bf16 Wc[16 * CCH];      // [m][t]   s*exp(acC-ac), 1KB
  __shared__ __align__(16) bf16 Pb[CCH * CCH];     // tril(QK^T) bf16, 2KB
  __shared__ float acS[CCH * 16];                  // [t][m] cum log-decay
  __shared__ float acC[16];
  __shared__ float IntraBuf[CCH * 16];

  const int tid = threadIdx.x;
  const int lane = tid & 31;
  const int w = tid >> 5;
  const int rowoff = (lane >> 4) << 3;

  v8f S[4];
#pragma unroll
  for (int j = 0; j < 4; ++j)
#pragma unroll
    for (int r = 0; r < 8; ++r) S[j][r] = 0.0f;

  for (int c = 0; c < TSEQ / CCH; ++c) {
    const int t0 = c * CCH;
    v8f inter = {};

    // Phase A: stage transposed K chunk; per-m cumulative decays -> acS/acC/Wc
#pragma unroll
    for (int rep = 0; rep < 8; ++rep) {
      int e8 = tid + rep * 128;                 // 1024 8-element segments
      int t = e8 >> 5;
      int dk0 = (e8 & 31) * 8;
      v8bf seg = *(const v8bf*)&kh[(size_t)(t0 + t) * DKE + dk0];
#pragma unroll
      for (int i = 0; i < 8; ++i) KcT[(dk0 + i) * CCH + t] = seg[i];
    }
    if (tid < 16) {
      float run = 0.0f;
      for (int t = 0; t < CCH; ++t) {
        run += flh[(size_t)(t0 + t) * DKE + tid];
        acS[t * 16 + tid] = run;
      }
      acC[tid] = run;
      for (int t = 0; t < CCH; ++t)
        Wc[tid * CCH + t] =
            (bf16)(sh[(size_t)(t0 + t) * DKE + tid] * expf(run - acS[t * 16 + tid]));
    }
    if (t0 + CCH < TSEQ) {   // prefetch next chunk streams into cache
      __builtin_prefetch(kh + (size_t)(t0 + CCH) * DKE + tid * 16, 0, 1);
      __builtin_prefetch(qh + (size_t)(t0 + CCH) * DKE + tid * 16, 0, 1);
    }
    __syncthreads();

    // Phase B: P = tril(Q K^T) per wave (B operand straight from global K);
    //          also scale state by exp(acC) and dump bf16 copy for Q@S0
    {
      const int tt = w >> 1, ti = w & 1;
      v8f p = {};
#pragma unroll
      for (int kk = 0; kk < 8; ++kk) {
        v16bf a  = load_a_bf(qh + (size_t)(t0 + tt * 16) * DKE, DKE, kk * 32);
        v16bf bm = load_b_bf(kh + (size_t)t0 * DKE, DKE, ti * 16, kk * 32);
        p = wmma_bf(a, bm, p);
      }
      const int i = ti * 16 + (lane & 15);
#pragma unroll
      for (int r = 0; r < 8; ++r) {
        const int t = tt * 16 + r + rowoff;
        Pb[t * CCH + i] = (i <= t) ? (bf16)p[r] : (bf16)0.0f;   // fused causal mask
      }
    }
    {
      float g = expf(acC[lane & 15]);   // column m = lane%16 for all state tiles
#pragma unroll
      for (int j = 0; j < 4; ++j) {
        int dkb = w * 64 + j * 16;
#pragma unroll
        for (int r = 0; r < 8; ++r) {
          S[j][r] *= g;
          S0c[(size_t)(lane & 15) * DKE + dkb + r + rowoff] = (bf16)S[j][r];
        }
      }
    }
    __syncthreads();

    // Phase C: waves 0-1 intra = tril(P) @ W ; waves 2-3 inter = Q @ S0_scaled
    if (w < 2) {
      const int tt = w;
      v8f intra = {};
      v16bf a  = load_a_bf(Pb + tt * 16 * CCH, CCH, 0);
      v16bf bm = load_b_bf(Wc, CCH, 0, 0);          // (t,m) from [m][t]
      intra = wmma_bf(a, bm, intra);
#pragma unroll
      for (int r = 0; r < 8; ++r)
        IntraBuf[(tt * 16 + r + rowoff) * 16 + (lane & 15)] = intra[r];
    } else {
      const int tt = w - 2;
#pragma unroll
      for (int kk = 0; kk < 8; ++kk) {
        v16bf a  = load_a_bf(qh + (size_t)(t0 + tt * 16) * DKE, DKE, kk * 32);
        v16bf bm = load_b_bf(S0c, DKE, 0, kk * 32); // (dk,m) from [m][dk]
        inter = wmma_bf(a, bm, inter);
      }
    }
    __syncthreads();

    // Phase D: combine & store ok; then accumulate state update S += K^T @ W
    if (w >= 2) {
      const int tt = w - 2;
#pragma unroll
      for (int r = 0; r < 8; ++r) {
        int t = tt * 16 + r + rowoff;
        int m = lane & 15;
        float val = (inter[r] + IntraBuf[t * 16 + m]) *
                    expf(acS[t * 16 + m] - acC[m]);
        okh[(size_t)(t0 + t) * DKE + m] = val;
      }
    }
    {
      v16bf bm = load_b_bf(Wc, CCH, 0, 0);
#pragma unroll
      for (int j = 0; j < 4; ++j) {
        int dkb = w * 64 + j * 16;
        v16bf a = load_a_bf(KcT + (size_t)dkb * CCH, CCH, 0);  // contiguous K^T
        S[j] = wmma_bf(a, bm, S[j]);
      }
    }
    __syncthreads();
  }
}

// ---------------- Kernel 3: softmax over slots (M=256) -------------------------
__global__ __launch_bounds__(128) void gsa_softmax(const float* __restrict__ ok,
                                                   bf16* __restrict__ qv)
{
  __shared__ float red[128];
  const int row = blockIdx.x;
  const int tid = threadIdx.x;
  const float* p = ok + (size_t)row * DKE;
  float a = p[tid], b = p[tid + 128];
  red[tid] = fmaxf(a, b);
  __syncthreads();
  for (int s2 = 64; s2 > 0; s2 >>= 1) {
    if (tid < s2) red[tid] = fmaxf(red[tid], red[tid + s2]);
    __syncthreads();
  }
  float mx = red[0];
  __syncthreads();
  float ea = expf(a - mx), eb = expf(b - mx);
  red[tid] = ea + eb;
  __syncthreads();
  for (int s2 = 64; s2 > 0; s2 >>= 1) {
    if (tid < s2) red[tid] += red[tid + s2];
    __syncthreads();
  }
  float inv = 1.0f / red[0];
  bf16* q = qv + (size_t)row * DKE;
  q[tid] = (bf16)(ea * inv);
  q[tid + 128] = (bf16)(eb * inv);
}

// ---------------- Kernel 4: pass-2 chunked GLA (state Hv[256 x 16] in accums) --
__global__ __launch_bounds__(128) void gsa_pass2(
    const bf16* __restrict__ qvg, const bf16* __restrict__ vg,
    const float* __restrict__ flg, const float* __restrict__ sg,
    float* __restrict__ og)
{
  const int bh = blockIdx.x >> 4;
  const int vbase = (blockIdx.x & 15) * 16;
  const bf16* qvh = qvg + (size_t)bh * TSEQ * DKE;
  const bf16* vh  = vg  + (size_t)bh * TSEQ * DKE + vbase;
  const float* flh = flg + (size_t)bh * TSEQ * DKE;
  const float* sh  = sg  + (size_t)bh * TSEQ * DKE;
  float* oh = og + (size_t)bh * TSEQ * DKE + vbase;

  __shared__ __align__(16) bf16 Wc2[DKE * CCH];    // [m][t] 16KB
  __shared__ __align__(16) bf16 WcT[CCH * DKE];    // [t][m] 16KB (transposed W)
  __shared__ __align__(16) bf16 Qt[CCH * DKE];     // [t][m] q~ = qv*exp(ac-acC)
  __shared__ __align__(16) bf16 Hc[16 * DKE];      // [v][m] scaled state
  __shared__ __align__(16) bf16 Vc[16 * CCH];      // [v][t]
  __shared__ __align__(16) bf16 Gb[CCH * CCH];     // tril(G) bf16
  __shared__ float acC2[DKE];

  const int tid = threadIdx.x;
  const int lane = tid & 31;
  const int w = tid >> 5;
  const int rowoff = (lane >> 4) << 3;

  v8f Hv[4];
#pragma unroll
  for (int j = 0; j < 4; ++j)
#pragma unroll
    for (int r = 0; r < 8; ++r) Hv[j][r] = 0.0f;

  for (int c = 0; c < TSEQ / CCH; ++c) {
    const int t0 = c * CCH;

    // Phase A: stage V chunk (col-major [v][t]); per-m decay scan -> Wc2/WcT/Qt
#pragma unroll
    for (int e = 0; e < 4; ++e) {
      int i = tid * 4 + e;
      int v = i >> 5, t = i & 31;
      Vc[v * CCH + t] = vh[(size_t)(t0 + t) * DKE + v];
    }
#pragma unroll
    for (int mm = 0; mm < 2; ++mm) {
      int m = tid + mm * 128;
      float run = 0.0f;
      for (int t = 0; t < CCH; ++t) run += flh[(size_t)(t0 + t) * DKE + m];
      acC2[m] = run;
      float run2 = 0.0f;
      for (int t = 0; t < CCH; ++t) {
        run2 += flh[(size_t)(t0 + t) * DKE + m];
        bf16 wv = (bf16)(sh[(size_t)(t0 + t) * DKE + m] * expf(run - run2));
        Wc2[m * CCH + t] = wv;
        WcT[t * DKE + m] = wv;
        Qt[t * DKE + m] =
            (bf16)((float)qvh[(size_t)(t0 + t) * DKE + m] * expf(run2 - run));
      }
    }
    if (t0 + CCH < TSEQ) {
      __builtin_prefetch(vh + (size_t)(t0 + CCH) * DKE + (tid & 31) * 8, 0, 1);
      __builtin_prefetch(flh + (size_t)(t0 + CCH) * DKE + tid * 2, 0, 1);
      __builtin_prefetch(sh + (size_t)(t0 + CCH) * DKE + tid * 2, 0, 1);
      __builtin_prefetch(qvh + (size_t)(t0 + CCH) * DKE + tid * 2, 0, 1);
    }
    __syncthreads();

    // Phase B: scale state rows by exp(acC2[m]) and dump bf16 copy [v][m]
#pragma unroll
    for (int j = 0; j < 4; ++j) {
      int mb = w * 64 + j * 16;
#pragma unroll
      for (int r = 0; r < 8; ++r) {
        int m = mb + r + rowoff;
        Hv[j][r] *= expf(acC2[m]);
        Hc[(size_t)(lane & 15) * DKE + m] = (bf16)Hv[j][r];
      }
    }
    __syncthreads();

    // Phase C: G = tril(Q~ @ W^T) (contraction over m=256), one tile per wave
    {
      const int tt = w >> 1, ti = w & 1;
      v8f g = {};
#pragma unroll
      for (int kk = 0; kk < 8; ++kk) {
        v16bf a  = load_a_bf(Qt + tt * 16 * DKE, DKE, kk * 32);
        v16bf bm = load_b_bf(WcT, DKE, ti * 16, kk * 32);  // contiguous [i][m]
        g = wmma_bf(a, bm, g);
      }
      const int i = ti * 16 + (lane & 15);
#pragma unroll
      for (int r = 0; r < 8; ++r) {
        const int t = tt * 16 + r + rowoff;
        Gb[t * CCH + i] = (i <= t) ? (bf16)g[r] : (bf16)0.0f;  // fused causal mask
      }
    }
    __syncthreads();

    // Phase D: o = Q~ @ H_scaled + tril(G) @ V ; state Hv += W^T @ V
    if (w < 2) {
      const int tt = w;
      v8f o = {};
#pragma unroll
      for (int kk = 0; kk < 8; ++kk) {
        v16bf a  = load_a_bf(Qt + tt * 16 * DKE, DKE, kk * 32);
        v16bf bm = load_b_bf(Hc, DKE, 0, kk * 32);  // (m,v) from [v][m]
        o = wmma_bf(a, bm, o);
      }
      {
        v16bf a  = load_a_bf(Gb + tt * 16 * CCH, CCH, 0);
        v16bf bm = load_b_bf(Vc, CCH, 0, 0);        // (t,v) from [v][t]
        o = wmma_bf(a, bm, o);
      }
#pragma unroll
      for (int r = 0; r < 8; ++r)
        oh[(size_t)(t0 + tt * 16 + r + rowoff) * DKE + (lane & 15)] = o[r];
    }
    {
      v16bf bm = load_b_bf(Vc, CCH, 0, 0);
#pragma unroll
      for (int j = 0; j < 4; ++j) {
        int mb = w * 64 + j * 16;
        v16bf a = load_a_bf(Wc2 + (size_t)mb * CCH, CCH, 0);   // contiguous W^T
        Hv[j] = wmma_bf(a, bm, Hv[j]);
      }
    }
    __syncthreads();
  }
}

// ---------------- Kernel 5: silu + RMSNorm(1024) * g_w -> bf16 -----------------
__global__ __launch_bounds__(256) void gsa_norm(const float* __restrict__ o2,
                                                const float* __restrict__ gw,
                                                bf16* __restrict__ yn)
{
  __shared__ float red[256];
  const int bt = blockIdx.x;
  const int b = bt >> 11, t = bt & 2047;
  const int tid = threadIdx.x;
  float vals[4];
  float ss = 0.0f;
#pragma unroll
  for (int e = 0; e < 4; ++e) {
    int j = tid + e * 256;
    int h = j >> 8, dv = j & 255;
    float x = o2[((size_t)(b * HH + h) * TSEQ + (size_t)t) * DKE + dv];
    x = silu_f(x);
    vals[e] = x;
    ss += x * x;
  }
  red[tid] = ss;
  __syncthreads();
  for (int s2 = 128; s2 > 0; s2 >>= 1) {
    if (tid < s2) red[tid] += red[tid + s2];
    __syncthreads();
  }
  float r = rsqrtf(red[0] * (1.0f / 1024.0f) + 1e-5f);
#pragma unroll
  for (int e = 0; e < 4; ++e) {
    int j = tid + e * 256;
    yn[(size_t)bt * DIM + j] = (bf16)(vals[e] * r * gw[j]);
  }
}

// ---------------- Kernel 6: output projection GEMM -----------------------------
__global__ __launch_bounds__(128) void gsa_ogemm(const bf16* __restrict__ yn,
                                                 const float* __restrict__ Wo,
                                                 float* __restrict__ out)
{
  const int tile = blockIdx.x * 4 + (threadIdx.x >> 5);
  const int rt = tile >> 6;            // 64 column tiles
  const int ct = tile & 63;
  const int rowBase = rt * 16, colBase = ct * 16;
  v8f acc = {};
#pragma unroll 4
  for (int kk = 0; kk < 32; ++kk) {
    v16bf a  = load_a_bf(yn + (size_t)rowBase * DIM, DIM, kk * 32);
    v16bf bm = load_b_f32(Wo, DIM, colBase, kk * 32);
    acc = wmma_bf(a, bm, acc);
  }
  const int lane = threadIdx.x & 31;
  const int rowoff = (lane >> 4) << 3;
#pragma unroll
  for (int r = 0; r < 8; ++r) {
    int bt = rowBase + r + rowoff;
    out[(size_t)bt * DIM + colBase + (lane & 15)] = acc[r];
  }
}

// ---------------- Host launcher -----------------------------------------------
extern "C" void kernel_launch(void* const* d_in, const int* in_sizes, int n_in,
                              void* d_out, int out_size, void* d_ws, size_t ws_size,
                              hipStream_t stream)
{
  (void)in_sizes; (void)n_in; (void)out_size; (void)ws_size;
  const float* hs = (const float*)d_in[0];
  const float* Wq = (const float*)d_in[1];
  const float* Wk = (const float*)d_in[2];
  const float* Wv = (const float*)d_in[3];
  const float* Wf = (const float*)d_in[4];
  const float* gw = (const float*)d_in[5];
  const float* Wo = (const float*)d_in[6];
  float* out = (float*)d_out;

  const size_t NE = (size_t)BB * HH * TSEQ * DKE;  // 4,194,304 per tensor
  char* ws = (char*)d_ws;
  bf16*  qb  = (bf16*)ws;  ws += NE * 2;
  bf16*  kb  = (bf16*)ws;  ws += NE * 2;
  bf16*  vb  = (bf16*)ws;  ws += NE * 2;
  bf16*  qvb = (bf16*)ws;  ws += NE * 2;
  float* flb = (float*)ws; ws += NE * 4;
  float* sb  = (float*)ws; ws += NE * 4;
  float* okb = (float*)ws; ws += NE * 4;
  float* o2b = (float*)ws; ws += NE * 4;
  bf16*  ynb = (bf16*)ws;  ws += (size_t)BB * TSEQ * DIM * 2;

  gsa_proj   <<<dim3(16384), dim3(128), 0, stream>>>(hs, Wq, Wk, Wv, Wf,
                                                     qb, kb, vb, flb, sb);
  gsa_pass1  <<<dim3(128),   dim3(128), 0, stream>>>(qb, kb, flb, sb, okb);
  gsa_softmax<<<dim3(16384), dim3(128), 0, stream>>>(okb, qvb);
  gsa_pass2  <<<dim3(128),   dim3(128), 0, stream>>>(qvb, vb, flb, sb, o2b);
  gsa_norm   <<<dim3(4096),  dim3(256), 0, stream>>>(o2b, gw, ynb);
  gsa_ogemm  <<<dim3(4096),  dim3(128), 0, stream>>>(ynb, Wo, out);
}
